// BaselineAttention_17806934409916
// MI455X (gfx1250) — matrix operand order repaired and checked by reading, not compiled, not run on hardware
//
#include <hip/hip_runtime.h>

typedef unsigned short u16;
typedef __attribute__((ext_vector_type(16))) __bf16 v16bf;
typedef __attribute__((ext_vector_type(8)))  float  v8f;
typedef __attribute__((ext_vector_type(4))) unsigned int u32x4;
typedef __attribute__((ext_vector_type(8))) int i32x8;
typedef __attribute__((ext_vector_type(4))) int i32x4;

union Frag16 { uint4 q[2]; v16bf v; };
union V4U    { uint4 q; u16 s[8]; };

__device__ __forceinline__ u16 f2bf(float f) {
  unsigned u = __float_as_uint(f);
  unsigned r = 0x7FFFu + ((u >> 16) & 1u);   // round-to-nearest-even
  return (u16)((u + r) >> 16);
}

// ---------------------------------------------------------------- TDM 2D tile load (bf16)
// Builds a D# descriptor (groups 0..3) and issues TENSOR_LOAD_TO_LDS.
// dims/strides in elements; tile stored packed row-by-row at lds_off.
__device__ __forceinline__ void tdm_load_2d_bf16(unsigned lds_off, const u16* gp,
                                                 unsigned tensor_d0, unsigned tensor_d1,
                                                 unsigned tile_d0, unsigned tile_d1,
                                                 unsigned stride0) {
  unsigned long long ga = (unsigned long long)gp;
  u32x4 g0;
  g0[0] = 1u;                                           // count=1 (valid user D#)
  g0[1] = lds_off;                                      // lds_addr (bytes)
  g0[2] = (unsigned)(ga & 0xFFFFFFFFu);                 // global_addr[31:0]
  g0[3] = (unsigned)((ga >> 32) & 0x01FFFFFFu)          // global_addr[56:32]
        | 0x80000000u;                                  // type=2 ("image")
  i32x8 g1;
  g1[0] = (int)(1u << 16);                              // data_size=1 -> 2 bytes
  g1[1] = (int)((tensor_d0 & 0xFFFFu) << 16);           // tensor_dim0[15:0]
  g1[2] = (int)((tensor_d0 >> 16) | ((tensor_d1 & 0xFFFFu) << 16));
  g1[3] = (int)((tensor_d1 >> 16) | (tile_d0 << 16));   // tile_dim0
  g1[4] = (int)(tile_d1 & 0xFFFFu);                     // tile_dim1 (tile_dim2=0)
  g1[5] = (int)stride0;                                 // tensor_dim0_stride[31:0]
  g1[6] = 0;                                            // stride0[47:32], stride1[15:0]
  g1[7] = 0;
  i32x4 g2 = {0, 0, 0, 0};
  i32x4 g3 = {0, 0, 0, 0};
  i32x8 g4 = {0, 0, 0, 0, 0, 0, 0, 0};
  __builtin_amdgcn_tensor_load_to_lds(g0, g1, g2, g3, g4, 0);
}

// ---------------------------------------------------------------- DPP butterfly reductions
// Reduce within each 16-lane half using xor-mask chain 15,7,3,1 (pure VALU, no LDS).
__device__ __forceinline__ float dpp_max16(float x) {
  int y;
  y = __builtin_amdgcn_update_dpp(0, __float_as_int(x), 0x140, 0xF, 0xF, true); // row_mirror
  x = fmaxf(x, __int_as_float(y));
  y = __builtin_amdgcn_update_dpp(0, __float_as_int(x), 0x141, 0xF, 0xF, true); // row_half_mirror
  x = fmaxf(x, __int_as_float(y));
  y = __builtin_amdgcn_update_dpp(0, __float_as_int(x), 0x1B, 0xF, 0xF, true);  // quad_perm(3,2,1,0)
  x = fmaxf(x, __int_as_float(y));
  y = __builtin_amdgcn_update_dpp(0, __float_as_int(x), 0xB1, 0xF, 0xF, true);  // quad_perm(1,0,3,2)
  x = fmaxf(x, __int_as_float(y));
  return x;
}
__device__ __forceinline__ float dpp_sum16(float x) {
  int y;
  y = __builtin_amdgcn_update_dpp(0, __float_as_int(x), 0x140, 0xF, 0xF, true);
  x += __int_as_float(y);
  y = __builtin_amdgcn_update_dpp(0, __float_as_int(x), 0x141, 0xF, 0xF, true);
  x += __int_as_float(y);
  y = __builtin_amdgcn_update_dpp(0, __float_as_int(x), 0x1B, 0xF, 0xF, true);
  x += __int_as_float(y);
  y = __builtin_amdgcn_update_dpp(0, __float_as_int(x), 0xB1, 0xF, 0xF, true);
  x += __int_as_float(y);
  return x;
}

// ---------------------------------------------------------------- convert
__global__ __launch_bounds__(256)
void cvt_f32_to_bf16(const float4* __restrict__ in, ushort4* __restrict__ out, int n4) {
  int i = blockIdx.x * blockDim.x + threadIdx.x;
  int stride = gridDim.x * blockDim.x;
  for (; i < n4; i += stride) {
    float4 f = in[i];
    ushort4 o;
    o.x = f2bf(f.x); o.y = f2bf(f.y); o.z = f2bf(f.z); o.w = f2bf(f.w);
    out[i] = o;
  }
}

// ---------------------------------------------------------------- GEMM: C = A * W^T + bias
// A: [M,K] bf16 row-major. W: [N,K] bf16 row-major (torch Linear weight).
// TDM-fed double-buffered LDS tiles; 8 waves, wave tile 16x64 (4 WMMA accumulators).
template<bool BF16OUT>
__global__ __launch_bounds__(256)
void gemm_xwT_bias(const u16* __restrict__ A, const u16* __restrict__ W,
                   const float* __restrict__ bias, void* __restrict__ Cout,
                   int M, int N, int K)
{
  __shared__ __align__(16) u16 As[2][64 * 32];
  __shared__ __align__(16) u16 Bs[2][128 * 32];

  const int t    = threadIdx.x;
  const int lane = t & 31, w = t >> 5;
  const int bm = blockIdx.y * 64, bn = blockIdx.x * 128;
  const int wm = (w >> 1) * 16,   wn = (w & 1) * 64;
  const int hl = lane >> 4,       ln = lane & 15;

  const unsigned as_off[2] = { (unsigned)(unsigned long long)&As[0][0],
                               (unsigned)(unsigned long long)&As[1][0] };
  const unsigned bs_off[2] = { (unsigned)(unsigned long long)&Bs[0][0],
                               (unsigned)(unsigned long long)&Bs[1][0] };

  v8f acc[4] = {};

  // prime the pipeline: tile for k0 = 0 into buffer 0 (TDM, async)
  if (w == 0) {
    tdm_load_2d_bf16(as_off[0], A + (size_t)bm * K, 32, 64,  32, 64,  (unsigned)K);
    tdm_load_2d_bf16(bs_off[0], W + (size_t)bn * K, 32, 128, 32, 128, (unsigned)K);
  }

  for (int k0 = 0; k0 < K; k0 += 32) {
    const int cur = (k0 >> 5) & 1;
    if (w == 0) {
      if (k0 + 32 < K) {
        // issue next tiles into the alternate buffer, then wait for current pair
        tdm_load_2d_bf16(as_off[cur ^ 1], A + (size_t)bm * K + k0 + 32, 32, 64,  32, 64,  (unsigned)K);
        tdm_load_2d_bf16(bs_off[cur ^ 1], W + (size_t)bn * K + k0 + 32, 32, 128, 32, 128, (unsigned)K);
        __builtin_amdgcn_s_wait_tensorcnt(2);
      } else {
        __builtin_amdgcn_s_wait_tensorcnt(0);
      }
    }
    __syncthreads();

    // A fragment 16x32: lanes 0-15 hold K 0..15, lanes 16-31 hold 16..31
    Frag16 a;
    a.q[0] = *(const uint4*)&As[cur][(wm + ln) * 32 + hl * 16];
    a.q[1] = *(const uint4*)&As[cur][(wm + ln) * 32 + hl * 16 + 8];
#pragma unroll
    for (int j = 0; j < 4; ++j) {
      // B fragment 32x16: lane n holds column n, 16 contiguous K values per half
      Frag16 b;
      int base = (wn + j * 16 + ln) * 32 + hl * 16;
      b.q[0] = *(const uint4*)&Bs[cur][base];
      b.q[1] = *(const uint4*)&Bs[cur][base + 8];
      acc[j] = __builtin_amdgcn_wmma_f32_16x16x32_bf16(
          false, a.v, false, b.v, (short)0, acc[j], false, false);
    }
    __syncthreads();
  }

#pragma unroll
  for (int r = 0; r < 8; ++r) {
    int row = bm + wm + hl * 8 + r;
#pragma unroll
    for (int j = 0; j < 4; ++j) {
      int col = bn + wn + j * 16 + ln;
      float v = acc[j][r] + bias[col];
      if (BF16OUT) ((u16*)Cout)[(size_t)row * N + col] = f2bf(v);
      else         ((float*)Cout)[(size_t)row * N + col] = v;
    }
  }
}

// ---------------------------------------------------------------- flash attention
// Grid: (16 q-blocks, 16 heads, 2 batch). Each block = 128 q rows, wave owns 16.
// K tile staged by TDM; V tile staged manually (transposed into LDS).
__global__ __launch_bounds__(256)
void attention_fa(const u16* __restrict__ Q, const u16* __restrict__ Kk,
                  const u16* __restrict__ V, u16* __restrict__ O)
{
  __shared__ __align__(16) u16 Ks[32 * 64];       // 32 keys x 64 d, row-major
  __shared__ __align__(16) u16 Vt[64 * 32];       // transposed: 64 d x 32 keys
  __shared__ __align__(16) u16 Ps[8][16 * 32];    // per-wave P scratch

  const int t = threadIdx.x, lane = t & 31, w = t >> 5;
  const int hl = lane >> 4, ln = lane & 15;
  const int b = blockIdx.z, h = blockIdx.y;
  const int q0 = blockIdx.x * 128 + w * 16;
  const size_t base = ((size_t)b * 2048) * 1024 + (size_t)h * 64;
  const unsigned ks_off = (unsigned)(unsigned long long)&Ks[0];

  // preload Q fragments (d = 0..31 and 32..63)
  Frag16 qa[2];
#pragma unroll
  for (int ds = 0; ds < 2; ++ds) {
    size_t off = base + (size_t)(q0 + ln) * 1024 + ds * 32;
    qa[ds].q[0] = *(const uint4*)&Q[off + hl * 16];
    qa[ds].q[1] = *(const uint4*)&Q[off + hl * 16 + 8];
  }

  float mrow[8], lrow[8];
#pragma unroll
  for (int r = 0; r < 8; ++r) { mrow[r] = -1e30f; lrow[r] = 0.f; }
  v8f acc[4] = {};

  const int krow = t >> 3, kcol = (t & 7) * 8;
  for (int kb = 0; kb < 2048; kb += 32) {
    // K tile 32x64 via TDM (row length 64, row stride 1024 elements)
    if (w == 0)
      tdm_load_2d_bf16(ks_off, Kk + base + (size_t)kb * 1024, 64, 32, 64, 32, 1024u);
    // V tile transposed, cooperative
    V4U vv;
    vv.q = *(const uint4*)&V[base + (size_t)(kb + krow) * 1024 + kcol];
#pragma unroll
    for (int j = 0; j < 8; ++j) Vt[(kcol + j) * 32 + krow] = vv.s[j];
    if (w == 0) __builtin_amdgcn_s_wait_tensorcnt(0);
    __syncthreads();

    // S = Q * K^T  (two key sub-tiles x two d-steps)
    v8f s0 = {}, s1 = {};
#pragma unroll
    for (int ds = 0; ds < 2; ++ds) {
      Frag16 k0f, k1f;
      int kb0 = (ln) * 64 + ds * 32 + hl * 16;
      int kb1 = (16 + ln) * 64 + ds * 32 + hl * 16;
      k0f.q[0] = *(const uint4*)&Ks[kb0]; k0f.q[1] = *(const uint4*)&Ks[kb0 + 8];
      k1f.q[0] = *(const uint4*)&Ks[kb1]; k1f.q[1] = *(const uint4*)&Ks[kb1 + 8];
      s0 = __builtin_amdgcn_wmma_f32_16x16x32_bf16(false, qa[ds].v, false, k0f.v,
                                                   (short)0, s0, false, false);
      s1 = __builtin_amdgcn_wmma_f32_16x16x32_bf16(false, qa[ds].v, false, k1f.v,
                                                   (short)0, s1, false, false);
    }
    s0 *= 0.125f;  // 1/sqrt(64)
    s1 *= 0.125f;

    // online softmax (row r lives in VGPR r across a 16-lane half); DPP reductions
#pragma unroll
    for (int r = 0; r < 8; ++r) {
      float tmax = dpp_max16(fmaxf(s0[r], s1[r]));
      float mnew = fmaxf(mrow[r], tmax);
      float p0 = __expf(s0[r] - mnew);
      float p1 = __expf(s1[r] - mnew);
      float corr = __expf(mrow[r] - mnew);
      mrow[r] = mnew;
      float rs = dpp_sum16(p0 + p1);
      lrow[r] = lrow[r] * corr + rs;
      int qr = hl * 8 + r;
      Ps[w][qr * 32 + ln]      = f2bf(p0);
      Ps[w][qr * 32 + 16 + ln] = f2bf(p1);
#pragma unroll
      for (int f = 0; f < 4; ++f) acc[f][r] *= corr;
    }

    // re-read P in A-layout (wave-private LDS, DS ops are in-order per wave)
    Frag16 pa;
    pa.q[0] = *(const uint4*)&Ps[w][ln * 32 + hl * 16];
    pa.q[1] = *(const uint4*)&Ps[w][ln * 32 + hl * 16 + 8];
#pragma unroll
    for (int f = 0; f < 4; ++f) {
      Frag16 vb;
      int vbase = (f * 16 + ln) * 32 + hl * 16;
      vb.q[0] = *(const uint4*)&Vt[vbase];
      vb.q[1] = *(const uint4*)&Vt[vbase + 8];
      acc[f] = __builtin_amdgcn_wmma_f32_16x16x32_bf16(false, pa.v, false, vb.v,
                                                       (short)0, acc[f], false, false);
    }
    __syncthreads();
  }

#pragma unroll
  for (int r = 0; r < 8; ++r) {
    int row = q0 + hl * 8 + r;
    float inv = 1.0f / lrow[r];
#pragma unroll
    for (int f = 0; f < 4; ++f)
      O[base + (size_t)row * 1024 + f * 16 + ln] = f2bf(acc[f][r] * inv);
  }
}

// ---------------------------------------------------------------- launch
extern "C" void kernel_launch(void* const* d_in, const int* in_sizes, int n_in,
                              void* d_out, int out_size, void* d_ws, size_t ws_size,
                              hipStream_t stream) {
  (void)in_sizes; (void)n_in; (void)out_size; (void)ws_size;
  const float* x  = (const float*)d_in[0];
  const float* WQ = (const float*)d_in[1];
  const float* bQ = (const float*)d_in[2];
  const float* WK = (const float*)d_in[3];
  const float* bK = (const float*)d_in[4];
  const float* WV = (const float*)d_in[5];
  const float* bV = (const float*)d_in[6];
  const float* WO = (const float*)d_in[7];
  const float* bO = (const float*)d_in[8];

  const int M = 4096, D = 1024;  // 2 * 2048 rows, d_model 1024
  char* ws = (char*)d_ws;
  u16* xb = (u16*)(ws);                      // 8 MB  x bf16
  u16* wq = (u16*)(ws + (8ull  << 20));      // 2 MB each weight
  u16* wk = (u16*)(ws + (10ull << 20));
  u16* wv = (u16*)(ws + (12ull << 20));
  u16* wo = (u16*)(ws + (14ull << 20));
  u16* Qb = (u16*)(ws + (16ull << 20));      // 8 MB each of Q,K,V,attn-out
  u16* Kb = (u16*)(ws + (24ull << 20));
  u16* Vb = (u16*)(ws + (32ull << 20));
  u16* Ob = (u16*)(ws + (40ull << 20));

  cvt_f32_to_bf16<<<1024, 256, 0, stream>>>((const float4*)x,  (ushort4*)xb, M * D / 4);
  cvt_f32_to_bf16<<<512,  256, 0, stream>>>((const float4*)WQ, (ushort4*)wq, D * D / 4);
  cvt_f32_to_bf16<<<512,  256, 0, stream>>>((const float4*)WK, (ushort4*)wk, D * D / 4);
  cvt_f32_to_bf16<<<512,  256, 0, stream>>>((const float4*)WV, (ushort4*)wv, D * D / 4);
  cvt_f32_to_bf16<<<512,  256, 0, stream>>>((const float4*)WO, (ushort4*)wo, D * D / 4);

  dim3 gg(D / 128, M / 64);
  gemm_xwT_bias<true><<<gg, 256, 0, stream>>>(xb, wq, bQ, Qb, M, D, D);
  gemm_xwT_bias<true><<<gg, 256, 0, stream>>>(xb, wk, bK, Kb, M, D, D);
  gemm_xwT_bias<true><<<gg, 256, 0, stream>>>(xb, wv, bV, Vb, M, D, D);

  attention_fa<<<dim3(16, 16, 2), 256, 0, stream>>>(Qb, Kb, Vb, Ob);

  gemm_xwT_bias<false><<<gg, 256, 0, stream>>>(Ob, wo, bO, d_out, M, D, D);
}
